// Encoder_79096117723484
// MI455X (gfx1250) — matrix operand order repaired and checked by reading, not compiled, not run on hardware
//
#include <hip/hip_runtime.h>
#include <hip/hip_bf16.h>

// ---------------------------------------------------------------------------
// CDNA5 / gfx1250 ConvGRU encoder, v4.
// All heavy convs are implicit GEMMs on v_wmma_f32_16x16x32_f16 (wave32),
// f16 NHWC activations, (r,s,ci)-ordered packed weights so each 32-wide K
// chunk (or 16-wide thread span) sits at a single filter tap -> one bounds
// check + one 32B vector load + one 32B LDS store per thread per chunk.
// 32co x 64px tiles (MT=2: two WMMA per wave per chunk, B-fragment reuse),
// double-buffered LDS, single barrier per chunk. Only stage-1's 1-channel
// 3x3 conv uses the generic scalar-gather path (K=9, one chunk).
// ---------------------------------------------------------------------------

typedef _Float16 half_t;
typedef __attribute__((ext_vector_type(16))) _Float16 v16h;
typedef __attribute__((ext_vector_type(8)))  _Float16 v8h;
typedef __attribute__((ext_vector_type(4)))  _Float16 v4h;
typedef __attribute__((ext_vector_type(8)))  float    v8f;

// ---------------------------- weight packing -------------------------------
// Flat K = (ci, r, s): src (Cout,Cin,3,3) -> f16 [Cout][Kpad]   (stage-1 ff)
__global__ void pack_w_flat_kernel(const float* __restrict__ src, half_t* __restrict__ dst,
                                   int Cout, int Kreal, int Kpad) {
    long i = (long)blockIdx.x * blockDim.x + threadIdx.x;
    long total = (long)Cout * Kpad;
    if (i >= total) return;
    int co = (int)(i / Kpad);
    int k  = (int)(i - (long)co * Kpad);
    dst[i] = (k < Kreal) ? (half_t)src[(long)co * Kreal + k] : (half_t)0.f;
}

// GRU 5x5: K ordered (r,s,ci): dst[co][rs*Ct + ci] = src[co][ci][r][s]
__global__ void pack_w_rsci_kernel(const float* __restrict__ src, half_t* __restrict__ dst,
                                   int Cout, int Ct) {
    long K = 25L * Ct;
    long i = (long)blockIdx.x * blockDim.x + threadIdx.x;
    long total = (long)Cout * K;
    if (i >= total) return;
    int co = (int)(i / K);
    int k  = (int)(i - (long)co * K);
    int rs = k / Ct;
    int ci = k - rs * Ct;
    dst[i] = (half_t)src[((long)co * Ct + ci) * 25 + rs];
}

// FF 3x3: K ordered (r,s,ci), zero-padded to Kpad
__global__ void pack_w_rsci3_kernel(const float* __restrict__ src, half_t* __restrict__ dst,
                                    int Cout, int Cin, int Kpad) {
    long i = (long)blockIdx.x * blockDim.x + threadIdx.x;
    long total = (long)Cout * Kpad;
    if (i >= total) return;
    int co = (int)(i / Kpad);
    int k  = (int)(i - (long)co * Kpad);
    int Kreal = 9 * Cin;
    if (k < Kreal) {
        int rs = k / Cin;
        int ci = k - rs * Cin;
        dst[i] = (half_t)src[((long)co * Cin + ci) * 9 + rs];
    } else {
        dst[i] = (half_t)0.f;
    }
}

__global__ void zero_f32_kernel(float* p, long n) {
    long i = (long)blockIdx.x * blockDim.x + threadIdx.x;
    if (i < n) p[i] = 0.f;
}
__global__ void zero_f16_kernel(half_t* p, long n) {
    long i = (long)blockIdx.x * blockDim.x + threadIdx.x;
    if (i < n) p[i] = (half_t)0.f;
}

// --------------------- GRU 5x5 conv (dominant GEMM) ------------------------
// Inputs f16 NHWC: ya=[frame][HW][C], hb=[n][HW][C]; channel concat [ya,hb].
// Weights f16 [Cout][25*2C], K ordered (r,s,ci). Output f32 NCHW.
template<int MT>
__global__ __launch_bounds__(128) void conv_gru_wmma(
    const half_t* __restrict__ ya, long strideA,
    const half_t* __restrict__ hb, long strideB,
    int C, const half_t* __restrict__ Wp, const float* __restrict__ bias,
    float* __restrict__ out, long strideOut,
    int H, int W, int Cout)
{
    const int HW  = H * W;
    const int n   = blockIdx.z;
    const int co0 = blockIdx.y * (MT * 16);
    const int p0  = blockIdx.x * 64;
    const int tid = threadIdx.x, lane = tid & 31, wave = tid >> 5;
    const int Ct  = 2 * C;
    const int K   = 25 * Ct;

    __shared__ __align__(32) half_t As[2][MT * 16 * 32];
    __shared__ __align__(32) half_t Bs[2][64 * 32];

    const int bp = tid & 63;             // pixel within tile (B staging)
    const int bk = (tid >> 6) * 16;      // 0 or 16 within k-chunk
    const int px = p0 + bp;              // HW % 64 == 0 -> always valid
    const int oh = px / W, ow = px - (px / W) * W;

    const half_t* ybase = ya + (long)n * strideA;
    const half_t* hbase = hb + (long)n * strideB;

    v8f acc[MT];
    #pragma unroll
    for (int mt = 0; mt < MT; ++mt) acc[mt] = (v8f){};

    int buf = 0;
    for (int rs = 0; rs < 25; ++rs) {
        const int r = rs / 5, s = rs - 5 * r;          // uniform (SALU)
        const int ih = oh + r - 2, iw = ow + s - 2;
        const bool valid = ((unsigned)ih < (unsigned)H) & ((unsigned)iw < (unsigned)W);
        const long spx = ((long)ih * W + iw) * C;      // used only if valid

        for (int cb = 0; cb < Ct; cb += 32) {
            const int kc = rs * Ct + cb;

            // ---- stage A: MT*512 halfs, MT*4 per thread, vector copy
            if constexpr (MT == 2) {
                int e = tid * 8, co = e >> 5, kk = e & 31;
                *((v8h*)&As[buf][e]) = *((const v8h*)(Wp + (long)(co0 + co) * K + kc + kk));
            } else {
                int e = tid * 4, co = e >> 5, kk = e & 31;
                *((v4h*)&As[buf][e]) = *((const v4h*)(Wp + (long)(co0 + co) * K + kc + kk));
            }
            if (kc + 32 < K)   // gfx1250 global_prefetch_b8 of next weight chunk
                __builtin_prefetch(Wp + (long)(co0 + ((tid >> 2) & (MT * 16 - 1))) * K + kc + 32, 0, 0);

            // ---- stage B: 16 contiguous channels per thread (32B load+store)
            {
                v16h tmp = {};
                if (valid) {
                    int ci0 = cb + bk;                 // multiple of 16; C%16==0
                    const half_t* src = (ci0 < C) ? (ybase + spx + ci0)
                                                  : (hbase + spx + (ci0 - C));
                    tmp = *(const v16h*)src;
                }
                *(v16h*)&Bs[buf][bp * 32 + bk] = tmp;
            }
            __syncthreads();   // single barrier per chunk (double buffer)

            // ---- fragments: contiguous k = 16*g + s in both A and B
            const int m = lane & 15, g = lane >> 4;
            v16h bfrag = *(const v16h*)&Bs[buf][(wave * 16 + m) * 32 + 16 * g];
            #pragma unroll
            for (int mt = 0; mt < MT; ++mt) {
                const half_t* arow = &As[buf][(mt * 16 + m) * 32];
                v8h lo = *(const v8h*)&arow[16 * g];
                v8h hi = *(const v8h*)&arow[16 * g + 8];
                v16h afrag = __builtin_shufflevector(lo, hi,
                    0,1,2,3,4,5,6,7,8,9,10,11,12,13,14,15);
                acc[mt] = __builtin_amdgcn_wmma_f32_16x16x32_f16(
                    false, afrag, false, bfrag, (short)0, acc[mt], false, false);
            }
            buf ^= 1;
        }
    }

    // ---- epilogue: f32 NCHW; C/D layout M = r + 8*(lane>=16), N = lane&15
    const int g   = lane >> 4;
    const int npx = p0 + wave * 16 + (lane & 15);
    float* ob = out + (long)n * strideOut + npx;
    #pragma unroll
    for (int mt = 0; mt < MT; ++mt) {
        #pragma unroll
        for (int r = 0; r < 8; ++r) {
            int co = co0 + mt * 16 + r + 8 * g;
            ob[(long)co * HW] = acc[mt][r] + bias[co];
        }
    }
}

// --------------- feedforward 3x3 stride-2 conv, f16 NHWC in ----------------
// Used for stages 2/3 (Cin = 16/64, multiple of 16 -> each 16-half staging
// span lies inside a single filter tap). Weights (r,s,ci). Output relu, NHWC.
template<int MT>
__global__ __launch_bounds__(128) void conv_ff_nhwc_wmma(
    const half_t* __restrict__ pin, long strideIn, int Cin,
    const half_t* __restrict__ Wp, const float* __restrict__ bias,
    half_t* __restrict__ out, long strideOut,
    int Hin, int Win, int Hout, int Wout, int Kpad, int Cout)
{
    constexpr int STR = 2, PAD = 1;
    const int n   = blockIdx.z;
    const int co0 = blockIdx.y * (MT * 16);
    const int p0  = blockIdx.x * 64;
    const int tid = threadIdx.x, lane = tid & 31, wave = tid >> 5;

    __shared__ __align__(32) half_t As[2][MT * 16 * 32];
    __shared__ __align__(32) half_t Bs[2][64 * 32];

    const int bp = tid & 63, bk = (tid >> 6) * 16;
    const int px = p0 + bp;                    // HWout % 64 == 0
    const int oh = px / Wout, ow = px - (px / Wout) * Wout;
    const half_t* base = pin + (long)n * strideIn;

    v8f acc[MT];
    #pragma unroll
    for (int mt = 0; mt < MT; ++mt) acc[mt] = (v8f){};

    int buf = 0;
    for (int kc = 0; kc < Kpad; kc += 32) {
        // ---- stage A
        if constexpr (MT == 2) {
            int e = tid * 8, co = e >> 5, kk = e & 31;
            *((v8h*)&As[buf][e]) = *((const v8h*)(Wp + (long)(co0 + co) * Kpad + kc + kk));
        } else {
            int e = tid * 4, co = e >> 5, kk = e & 31;
            *((v4h*)&As[buf][e]) = *((const v4h*)(Wp + (long)(co0 + co) * Kpad + kc + kk));
        }
        // ---- stage B: one tap + 16 contiguous channels per thread
        {
            v16h tmp = {};
            int seg = kc + bk;
            int rs  = seg / Cin;               // single division per chunk
            int ci0 = seg - rs * Cin;
            if (rs < 9) {
                int r = rs / 3, s = rs - 3 * r;
                int ih = oh * STR + r - PAD, iw = ow * STR + s - PAD;
                if ((unsigned)ih < (unsigned)Hin && (unsigned)iw < (unsigned)Win)
                    tmp = *(const v16h*)(base + ((long)ih * Win + iw) * Cin + ci0);
            }
            *(v16h*)&Bs[buf][bp * 32 + bk] = tmp;
        }
        __syncthreads();

        const int m = lane & 15, g = lane >> 4;
        v16h bfrag = *(const v16h*)&Bs[buf][(wave * 16 + m) * 32 + 16 * g];
        #pragma unroll
        for (int mt = 0; mt < MT; ++mt) {
            const half_t* arow = &As[buf][(mt * 16 + m) * 32];
            v8h lo = *(const v8h*)&arow[16 * g];
            v8h hi = *(const v8h*)&arow[16 * g + 8];
            v16h afrag = __builtin_shufflevector(lo, hi,
                0,1,2,3,4,5,6,7,8,9,10,11,12,13,14,15);
            acc[mt] = __builtin_amdgcn_wmma_f32_16x16x32_f16(
                false, afrag, false, bfrag, (short)0, acc[mt], false, false);
        }
        buf ^= 1;
    }

    // ---- epilogue: relu, f16 NHWC, one b128 store per sub-tile
    const int g   = lane >> 4;
    const int npx = p0 + wave * 16 + (lane & 15);
    half_t* ob = out + (long)n * strideOut + (long)npx * Cout;
    #pragma unroll
    for (int mt = 0; mt < MT; ++mt) {
        v8h o8;
        #pragma unroll
        for (int r = 0; r < 8; ++r) {
            int co = co0 + mt * 16 + 8 * g + r;    // acc[mt][r]: M = mt*16 + r + 8g
            o8[r] = (half_t)fmaxf(acc[mt][r] + bias[co], 0.f);
        }
        *(v8h*)(ob + co0 + mt * 16 + 8 * g) = o8;
    }
}

// ----------- stage-1 feedforward 3x3 conv (f32 NCHW in, Cin=1) -------------
__global__ __launch_bounds__(128) void conv_ff_wmma(
    const float* __restrict__ pin, long strideIn, int Cin,
    const half_t* __restrict__ Wp, const float* __restrict__ bias,
    half_t* __restrict__ out, long strideOut,
    int Hin, int Win, int Hout, int Wout, int stride, int Kpad, int Cout)
{
    constexpr int KS = 3, KSQ = 9, PAD = 1;
    const int n   = blockIdx.z;
    const int co0 = blockIdx.y * 16;
    const int p0  = blockIdx.x * 64;
    const int tid = threadIdx.x, lane = tid & 31, wave = tid >> 5;
    const int Kreal = Cin * KSQ;

    __shared__ __align__(32) half_t As[2][512];
    __shared__ __align__(32) half_t Bs[2][2048];

    const int bp = tid & 63, bk = (tid >> 6) * 16;
    const int px = p0 + bp;                    // HWout % 64 == 0
    const int oh = px / Wout, ow = px - (px / Wout) * Wout;
    const float* base = pin + (long)n * strideIn;

    v8f acc = {};
    int buf = 0;
    for (int kc = 0; kc < Kpad; kc += 32) {
        {
            int e = tid * 4, co = e >> 5, kk = e & 31;
            *((v4h*)&As[buf][e]) = *((const v4h*)(Wp + (long)(co0 + co) * Kpad + kc + kk));
        }
        {
            v16h tmp = {};
            #pragma unroll            // full unroll: constant lane indices
            for (int j = 0; j < 16; ++j) {
                int k = kc + bk + j;
                float v = 0.f;
                if (k < Kreal) {
                    int ci = k / KSQ, rem = k - ci * KSQ;
                    int r = rem / KS, s = rem - r * KS;
                    int ih = oh * stride + r - PAD, iw = ow * stride + s - PAD;
                    if ((unsigned)ih < (unsigned)Hin && (unsigned)iw < (unsigned)Win)
                        v = base[((long)ci * Hin + ih) * Win + iw];
                }
                tmp[j] = (half_t)v;
            }
            *(v16h*)&Bs[buf][bp * 32 + bk] = tmp;
        }
        __syncthreads();

        const int m = lane & 15, g = lane >> 4;
        const half_t* arow = &As[buf][m * 32];
        v8h lo = *(const v8h*)&arow[16 * g];
        v8h hi = *(const v8h*)&arow[16 * g + 8];
        v16h afrag = __builtin_shufflevector(lo, hi,
            0,1,2,3,4,5,6,7,8,9,10,11,12,13,14,15);
        v16h bfrag = *(const v16h*)&Bs[buf][(wave * 16 + m) * 32 + 16 * g];
        acc = __builtin_amdgcn_wmma_f32_16x16x32_f16(
            false, afrag, false, bfrag, (short)0, acc, false, false);
        buf ^= 1;
    }

    const int g   = lane >> 4;
    const int npx = p0 + wave * 16 + (lane & 15);
    v8h o8;
    #pragma unroll
    for (int r = 0; r < 8; ++r) {
        int co = co0 + 8 * g + r;              // acc[r] is M = r + 8*g
        o8[r] = (half_t)fmaxf(acc[r] + bias[co], 0.f);
    }
    *(v8h*)(out + (long)n * strideOut + (long)npx * Cout + co0 + 8 * g) = o8;
}

// ------------------------------- GroupNorm ---------------------------------
__global__ __launch_bounds__(256) void gn_stats_kernel(
    const float* __restrict__ x, float* __restrict__ stats, int C, int HW, int G)
{
    int ng = blockIdx.x;
    int n = ng / G, g = ng - n * G;
    int cpg = C / G;
    const float* p = x + ((long)n * C + (long)g * cpg) * HW;
    long cnt = (long)cpg * HW;
    float s = 0.f, s2 = 0.f;
    for (long i = threadIdx.x; i < cnt; i += 256) { float v = p[i]; s += v; s2 += v * v; }
    __shared__ float sh[256], sh2[256];
    sh[threadIdx.x] = s; sh2[threadIdx.x] = s2;
    __syncthreads();
    for (int off = 128; off > 0; off >>= 1) {
        if (threadIdx.x < (unsigned)off) {
            sh[threadIdx.x]  += sh[threadIdx.x + off];
            sh2[threadIdx.x] += sh2[threadIdx.x + off];
        }
        __syncthreads();
    }
    if (threadIdx.x == 0) {
        float mean = sh[0] / (float)cnt;
        float var  = sh2[0] / (float)cnt - mean * mean;
        stats[ng * 2]     = mean;
        stats[ng * 2 + 1] = rsqrtf(var + 1e-5f);
    }
}

// act: 0 none, 1 sigmoid, 2 tanh
__global__ void gn_apply_kernel(const float* __restrict__ x, float* __restrict__ y,
                                const float* __restrict__ stats,
                                const float* __restrict__ gamma, const float* __restrict__ beta,
                                int C, int HW, int G, long total, int act)
{
    long i = (long)blockIdx.x * blockDim.x + threadIdx.x;
    if (i >= total) return;
    int cpg = C / G;
    long chw = (long)C * HW;
    int n = (int)(i / chw);
    int c = (int)((i / HW) % C);
    int g = c / cpg;
    float mean = stats[(n * G + g) * 2];
    float rstd = stats[(n * G + g) * 2 + 1];
    float v = (x[i] - mean) * rstd * gamma[c] + beta[c];
    if (act == 1)      v = 1.f / (1.f + __expf(-v));
    else if (act == 2) v = tanhf(v);
    y[i] = v;
}

// ------------------------------- GRU pointwise -----------------------------
// rh16 (f16 NHWC) = sigmoid(r) * h   (r = gates channels [C,2C), f32 NCHW)
__global__ void rh_mul_kernel(const float* __restrict__ gates, const float* __restrict__ h,
                              half_t* __restrict__ rh16, int C, int HW, long total)
{
    long i = (long)blockIdx.x * blockDim.x + threadIdx.x;
    if (i >= total) return;
    long chw = (long)C * HW;
    int n = (int)(i / chw);
    long rest = i - (long)n * chw;
    int c = (int)(rest / HW);
    int hw = (int)(rest - (long)c * HW);
    float r = gates[(long)n * 2 * chw + chw + rest];
    rh16[(long)n * chw + (long)hw * C + c] = (half_t)(r * h[i]);
}

// h = (1-z)*h + z*ht ; mirrors: h16 (NHWC), o16 (NHWC per (n*T+t) frame);
// o[:,t] = h (f32 NCHW, d_out)
__global__ void gru_update_kernel(const float* __restrict__ gates, const float* __restrict__ ht,
                                  float* __restrict__ h, half_t* __restrict__ h16,
                                  float* __restrict__ o, half_t* __restrict__ o16,
                                  int C, int HW, int T, int t, long total)
{
    long i = (long)blockIdx.x * blockDim.x + threadIdx.x;
    if (i >= total) return;
    long chw = (long)C * HW;
    int n = (int)(i / chw);
    long rest = i - (long)n * chw;
    int c = (int)(rest / HW);
    int hw = (int)(rest - (long)c * HW);
    float z  = gates[(long)n * 2 * chw + rest];
    float hn = (1.f - z) * h[i] + z * ht[i];
    h[i] = hn;
    h16[(long)n * chw + (long)hw * C + c] = (half_t)hn;
    o16[(long)(n * T + t) * chw + (long)hw * C + c] = (half_t)hn;
    o[(long)(n * T + t) * chw + rest] = hn;
}

// ------------------------------- host driver -------------------------------
static inline long rup(long v, long a) { return (v + a - 1) / a * a; }
static inline dim3 g1d(long n, int bs) { return dim3((unsigned)((n + bs - 1) / bs)); }

static void launch_gru_conv(hipStream_t st, const half_t* ya, long sA,
                            const half_t* hb, long sB, int C,
                            const half_t* Wp, const float* bias,
                            float* out, long sOut, int H, int W, int Cout, int B)
{
    if (Cout % 32 == 0) {
        dim3 grid((H * W) / 64, Cout / 32, B);
        conv_gru_wmma<2><<<grid, 128, 0, st>>>(ya, sA, hb, sB, C, Wp, bias, out, sOut, H, W, Cout);
    } else {
        dim3 grid((H * W) / 64, Cout / 16, B);
        conv_gru_wmma<1><<<grid, 128, 0, st>>>(ya, sA, hb, sB, C, Wp, bias, out, sOut, H, W, Cout);
    }
}

extern "C" void kernel_launch(void* const* d_in, const int* in_sizes, int n_in,
                              void* d_out, int out_size, void* d_ws, size_t ws_size,
                              hipStream_t stream)
{
    (void)in_sizes; (void)n_in; (void)out_size; (void)ws_size;
    const int B = 16, T = 16;

    struct SP { int Cin, C, Hin, Win, Hout, Wout, stride, iw; long o_off, h_off; };
    SP sp[3] = {
        { 1, 16, 64, 64, 64, 64, 1,  1, 0, 0},
        {16, 64, 64, 64, 32, 32, 2, 11, 0, 0},
        {64, 96, 32, 32, 16, 16, 2, 21, 0, 0},
    };
    long off = 0;
    for (int s = 0; s < 3; ++s) { sp[s].o_off = off; off += (long)B * T * sp[s].C * sp[s].Hout * sp[s].Wout; }
    for (int s = 0; s < 3; ++s) { sp[s].h_off = off; off += (long)B * sp[s].C * sp[s].Hout * sp[s].Wout; }

    const float* x = (const float*)d_in[0];
    float* outp = (float*)d_out;

    // ---- workspace bump allocator (~92 MB total) ----
    char* ws = (char*)d_ws;
    size_t wo = 0;
    auto allocf = [&](long nE) -> float*  { float*  p = (float*) (ws + wo); wo += (size_t)rup(nE * 4, 256); return p; };
    auto alloch = [&](long nE) -> half_t* { half_t* p = (half_t*)(ws + wo); wo += (size_t)rup(nE * 2, 256); return p; };

    half_t* y16   = alloch(16777216);  // max B*T*C*HW, f16 NHWC (post-ff y)
    half_t* o16   = alloch(16777216);  // f16 NHWC mirror of previous-stage o
    float*  gates = allocf(2097152);   // max B*2C*HW, f32 NCHW
    float*  htbuf = allocf(1048576);   // max B*C*HW
    float*  hbuf  = allocf(1048576);   // f32 NCHW hidden
    half_t* h16   = alloch(1048576);   // f16 NHWC hidden mirror
    half_t* rh16  = alloch(1048576);   // f16 NHWC r*h
    float*  stats = allocf(1024);

    half_t* wpc[3]; half_t* wp1[3]; half_t* wp2[3];
    int kpc[3];
    for (int s = 0; s < 3; ++s) {
        kpc[s] = (int)rup(sp[s].Cin * 9, 32);
        wpc[s] = alloch((long)sp[s].C * kpc[s]);
        wp1[s] = alloch((long)2 * sp[s].C * 50L * sp[s].C);   // 2C x 25*2C
        wp2[s] = alloch((long)sp[s].C * 50L * sp[s].C);       //  C x 25*2C
    }

    for (int s = 0; s < 3; ++s) {
        const SP& P = sp[s];
        const int C = P.C, HW = P.Hout * P.Wout;
        const long chw  = (long)C * HW;
        const long tot1 = (long)B * chw;
        const long tot2 = 2 * tot1;
        const int G1 = (2 * C) / 8, G2 = C / 8;
        const int Ct = 2 * C;

        const float* cw  = (const float*)d_in[P.iw + 0];
        const float* cb  = (const float*)d_in[P.iw + 1];
        const float* w1  = (const float*)d_in[P.iw + 2];
        const float* b1  = (const float*)d_in[P.iw + 3];
        const float* g1g = (const float*)d_in[P.iw + 4];
        const float* g1b = (const float*)d_in[P.iw + 5];
        const float* w2  = (const float*)d_in[P.iw + 6];
        const float* b2  = (const float*)d_in[P.iw + 7];
        const float* g2g = (const float*)d_in[P.iw + 8];
        const float* g2b = (const float*)d_in[P.iw + 9];

        // pack weights
        if (s == 0)
            pack_w_flat_kernel<<<g1d((long)C * kpc[s], 256), 256, 0, stream>>>(cw, wpc[s], C, P.Cin * 9, kpc[s]);
        else
            pack_w_rsci3_kernel<<<g1d((long)C * kpc[s], 256), 256, 0, stream>>>(cw, wpc[s], C, P.Cin, kpc[s]);
        pack_w_rsci_kernel<<<g1d((long)2 * C * 25 * Ct, 256), 256, 0, stream>>>(w1, wp1[s], 2 * C, Ct);
        pack_w_rsci_kernel<<<g1d((long)C * 25 * Ct, 256), 256, 0, stream>>>(w2, wp2[s], C, Ct);

        // TimeDistributed feedforward: y = relu(conv3x3(stage input)) -> f16 NHWC
        if (s == 0) {
            dim3 grid(HW / 64, C / 16, B * T);
            conv_ff_wmma<<<grid, 128, 0, stream>>>(
                x, (long)P.Cin * P.Hin * P.Win, P.Cin, wpc[s], cb,
                y16, chw, P.Hin, P.Win, P.Hout, P.Wout, P.stride, kpc[s], C);
        } else {
            // input: o16 (f16 NHWC mirror of previous stage's o); Cout 64/96 -> MT=2
            dim3 grid(HW / 64, C / 32, B * T);
            conv_ff_nhwc_wmma<2><<<grid, 128, 0, stream>>>(
                o16, (long)P.Cin * P.Hin * P.Win, P.Cin, wpc[s], cb,
                y16, chw, P.Hin, P.Win, P.Hout, P.Wout, kpc[s], C);
        }

        // h0 = 0 (f32 NCHW + f16 NHWC mirror)
        zero_f32_kernel<<<g1d(tot1, 256), 256, 0, stream>>>(hbuf, tot1);
        zero_f16_kernel<<<g1d(tot1, 256), 256, 0, stream>>>(h16, tot1);

        float* o_ptr = outp + P.o_off;
        for (int t = 0; t < T; ++t) {
            const half_t* yt = y16 + (long)t * chw;   // frame n -> (n*T+t)
            const long ysA = (long)T * chw;

            // gates = GN1(conv5x5([y_t, h])), sigmoid
            launch_gru_conv(stream, yt, ysA, h16, chw, C, wp1[s], b1,
                            gates, 2 * chw, P.Hout, P.Wout, 2 * C, B);
            gn_stats_kernel<<<dim3(B * G1), 256, 0, stream>>>(gates, stats, 2 * C, HW, G1);
            gn_apply_kernel<<<g1d(tot2, 256), 256, 0, stream>>>(gates, gates, stats, g1g, g1b,
                                                                2 * C, HW, G1, tot2, 1);
            // rh = r * h (f16 NHWC)
            rh_mul_kernel<<<g1d(tot1, 256), 256, 0, stream>>>(gates, hbuf, rh16, C, HW, tot1);

            // ht = tanh(GN2(conv5x5([y_t, rh])))
            launch_gru_conv(stream, yt, ysA, rh16, chw, C, wp2[s], b2,
                            htbuf, chw, P.Hout, P.Wout, C, B);
            gn_stats_kernel<<<dim3(B * G2), 256, 0, stream>>>(htbuf, stats, C, HW, G2);
            gn_apply_kernel<<<g1d(tot1, 256), 256, 0, stream>>>(htbuf, htbuf, stats, g2g, g2b,
                                                                C, HW, G2, tot1, 2);
            // h = (1-z)*h + z*ht ; o[:,t] = h (+ NHWC mirrors)
            gru_update_kernel<<<g1d(tot1, 256), 256, 0, stream>>>(gates, htbuf, hbuf, h16,
                                                                  o_ptr, o16, C, HW, T, t, tot1);
        }

        hipMemcpyAsync(outp + P.h_off, hbuf, (size_t)tot1 * sizeof(float),
                       hipMemcpyDeviceToDevice, stream);
    }
}
